// FlashLongconv_17824114279097
// MI455X (gfx1250) — matrix operand order repaired and checked by reading, not compiled, not run on hardware
//
#include <hip/hip_runtime.h>
#include <hip/hip_bf16.h>
#include <math.h>

#define B_  2
#define H_  2048
#define L_  4096
#define LK_ 128
#define O_  4096      // 2H
#define LAM 0.003f

typedef __bf16 v16bf __attribute__((ext_vector_type(16)));
typedef __bf16 v8bf  __attribute__((ext_vector_type(8)));
typedef float  v8f   __attribute__((ext_vector_type(8)));

static __device__ __forceinline__ __bf16 f2bf(float f) {
  unsigned u = __builtin_bit_cast(unsigned, f);
  u += 0x7FFFu + ((u >> 16) & 1u);           // round-to-nearest-even
  unsigned short s = (unsigned short)(u >> 16);
  return __builtin_bit_cast(__bf16, s);
}

// async global->LDS copy of 16 bytes (ASYNCcnt-tracked, CDNA5)
static __device__ __forceinline__ void async_b128(unsigned lds_off, const void* g) {
  asm volatile("global_load_async_to_lds_b128 %0, %1, off"
               :: "v"(lds_off), "v"((unsigned long long)(uintptr_t)g)
               : "memory");
}

// ---------------------------------------------------------------- kernel 0:
// W_out (f32 [O_, H_]) -> bf16 workspace
__global__ __launch_bounds__(256) void wcvt_kernel(const float* __restrict__ W,
                                                   __bf16* __restrict__ Wb) {
  size_t i = ((size_t)blockIdx.x * 256 + threadIdx.x) * 4;
  float4 w = *(const float4*)(W + i);
  Wb[i + 0] = f2bf(w.x);
  Wb[i + 1] = f2bf(w.y);
  Wb[i + 2] = f2bf(w.z);
  Wb[i + 3] = f2bf(w.w);
}

// ---------------------------------------------------------------- kernel 1:
// depthwise 128-tap causal conv (== zero-padded rfft/irfft) + D*u + exact
// GELU; writes Y TRANSPOSED as bf16 [B, L, H] so the GEMM can stage tiles
// with contiguous async b128 copies (no LDS transpose in the hot loop).
#define HT  16
#define LT2 256
#define UW  (LT2 + LK_)   // 384 floats per row (uses 383)
__global__ __launch_bounds__(256) void conv_gelu_kernel(
    const float* __restrict__ u, const float* __restrict__ kern,
    const float* __restrict__ Dv, __bf16* __restrict__ Yt) {
  __shared__ float su[HT][UW];    // 24 KiB
  __shared__ float sk[HT][LK_];   //  8 KiB
  const int l0  = blockIdx.x * LT2;
  const int h0  = blockIdx.y * HT;
  const int b   = blockIdx.z;
  const int tid = threadIdx.x;

  for (int i = tid; i < HT * LK_; i += 256) {   // soft-thresholded taps
    int hh = i >> 7, j = i & (LK_ - 1);
    float kv = kern[(h0 + hh) * LK_ + j];
    float a  = fabsf(kv) - LAM;
    sk[hh][j] = a > 0.f ? copysignf(a, kv) : 0.f;
  }
  for (int i = tid; i < HT * (LT2 + LK_ - 1); i += 256) {  // u + left halo
    int hh = i / (LT2 + LK_ - 1);
    int x  = i - hh * (LT2 + LK_ - 1);
    int l  = l0 - (LK_ - 1) + x;
    su[hh][x] = (l >= 0) ? u[((size_t)(b * H_ + h0 + hh)) * (size_t)L_ + l] : 0.f;
  }
  __syncthreads();

  const int li = tid;                       // this thread's column l0+li
  alignas(16) __bf16 ov[HT];
#pragma unroll
  for (int hh = 0; hh < HT; ++hh) {
    float acc = Dv[h0 + hh] * su[hh][LK_ - 1 + li];
#pragma unroll 8
    for (int j = 0; j < LK_; ++j)
      acc = fmaf(sk[hh][j], su[hh][LK_ - 1 + li - j], acc);
    float g = 0.5f * acc * (1.f + erff(acc * 0.70710678118654752f));  // exact GELU
    ov[hh] = f2bf(g);
  }
  __bf16* yp = Yt + ((size_t)b * L_ + l0 + li) * (size_t)H_ + h0;  // 32B aligned
  *(v8bf*)yp       = *(const v8bf*)&ov[0];
  *(v8bf*)(yp + 8) = *(const v8bf*)&ov[8];
}

// ---------------------------------------------------------------- kernel 2:
// fused GEMM (bf16 WMMA, f32 accum) + bias + GLU, async double-buffered LDS.
// Block: output rows [m0,m0+64) ('a') and [m0+2048,m0+2112) ('g') x 128 cols.
// 8 waves (2 x 4), each wave 32x32 of both halves = 8 WMMA accum tiles.
#define BM  64
#define BN  128
#define BK  32
#define LDA 40      // 80 B = 20 dwords -> conflict-free, 16B-aligned chunks
#define LDB 40
#define KT  (H_ / BK)   // 64 K-tiles

__global__ __launch_bounds__(256) void gemm_glu_kernel(
    const __bf16* __restrict__ Wb,   // [O_, H_]
    const __bf16* __restrict__ Yt,   // [B_, L_, H_]  (transposed)
    const float*  __restrict__ bout, // [O_]
    float*        __restrict__ out)  // [B_, H_, L_]
{
  __shared__ __align__(16) __bf16 As[2][2 * BM][LDA]; // [buf][row 0..63 'a', 64..127 'g']
  __shared__ __align__(16) __bf16 Bs[2][BN][LDB];     // [buf][n][k]

  const int n0   = blockIdx.x * BN;
  const int m0   = blockIdx.y * BM;
  const int b    = blockIdx.z;
  const int tid  = threadIdx.x;
  const int lane = tid & 31;
  const int wid  = tid >> 5;
  const int wm   = wid & 1;    // 32-row slice of BM
  const int wn   = wid >> 1;   // 32-col slice of BN

  v8f ca[2][2], cg[2][2];
#pragma unroll
  for (int i = 0; i < 2; ++i)
#pragma unroll
    for (int j = 0; j < 2; ++j)
#pragma unroll
      for (int r = 0; r < 8; ++r) { ca[i][j][r] = 0.f; cg[i][j][r] = 0.f; }

  // staging: thread -> (row, 16-elem half of BK); both tiles are k-contiguous
  const int srow  = tid >> 1;           // 0..127
  const int shalf = tid & 1;            // 0..1
  const int oa    = (srow < BM) ? (m0 + srow) : (H_ + m0 + (srow - BM));
  const __bf16* aptr = Wb + (size_t)oa * H_ + shalf * 16;
  const __bf16* bptr = Yt + ((size_t)b * L_ + n0 + srow) * (size_t)H_ + shalf * 16;
  const unsigned aoff0 = (unsigned)(uintptr_t)&As[0][srow][shalf * 16];
  const unsigned boff0 = (unsigned)(uintptr_t)&Bs[0][srow][shalf * 16];
  const unsigned abufstride = (unsigned)(2 * BM * LDA * sizeof(__bf16));
  const unsigned bbufstride = (unsigned)(BN * LDB * sizeof(__bf16));

  const int mrow = lane & 15;
  const int kb   = (lane >> 4) * 8;   // A frag: K 0..7 / 8..15 by lane half
  const int koff = (lane >> 4) * 8;   // B frag: K 0..7 / 8..15 by lane half

  // preload tile 0
  async_b128(aoff0,      aptr);
  async_b128(aoff0 + 16, aptr + 8);
  async_b128(boff0,      bptr);
  async_b128(boff0 + 16, bptr + 8);

  for (int t = 0; t < KT; ++t) {
    const int buf = t & 1;
    if (t + 1 < KT) {                          // issue next tile into other buf
      const int k1 = (t + 1) * BK;
      const unsigned nb = (t + 1) & 1;
      async_b128(aoff0 + nb * abufstride,      aptr + k1);
      async_b128(aoff0 + nb * abufstride + 16, aptr + k1 + 8);
      async_b128(boff0 + nb * bbufstride,      bptr + k1);
      async_b128(boff0 + nb * bbufstride + 16, bptr + k1 + 8);
      asm volatile("s_wait_asynccnt 0x4" ::: "memory");  // tile t landed (in-order)
    } else {
      asm volatile("s_wait_asynccnt 0x0" ::: "memory");
    }
    __syncthreads();

    // fragments per documented wave32 VGPR layouts
    v16bf af[2], gf[2], bfr[2];
#pragma unroll
    for (int i = 0; i < 2; ++i) {
      const int ra = wm * 32 + i * 16 + mrow;
      v8bf lo  = *(const v8bf*)&As[buf][ra][kb];
      v8bf hi  = *(const v8bf*)&As[buf][ra][kb + 16];
      af[i] = __builtin_shufflevector(lo, hi, 0,1,2,3,4,5,6,7,8,9,10,11,12,13,14,15);
      v8bf glo = *(const v8bf*)&As[buf][BM + ra][kb];
      v8bf ghi = *(const v8bf*)&As[buf][BM + ra][kb + 16];
      gf[i] = __builtin_shufflevector(glo, ghi, 0,1,2,3,4,5,6,7,8,9,10,11,12,13,14,15);
    }
#pragma unroll
    for (int j = 0; j < 2; ++j) {
      const int n = wn * 32 + j * 16 + (lane & 15);
      v8bf blo = *(const v8bf*)&Bs[buf][n][koff];
      v8bf bhi = *(const v8bf*)&Bs[buf][n][koff + 16];
      bfr[j] = __builtin_shufflevector(blo, bhi, 0,1,2,3,4,5,6,7,8,9,10,11,12,13,14,15);
    }
#pragma unroll
    for (int i = 0; i < 2; ++i)
#pragma unroll
      for (int j = 0; j < 2; ++j) {
        ca[i][j] = __builtin_amdgcn_wmma_f32_16x16x32_bf16(
            false, af[i], false, bfr[j], (short)0, ca[i][j], false, false);
        cg[i][j] = __builtin_amdgcn_wmma_f32_16x16x32_bf16(
            false, gf[i], false, bfr[j], (short)0, cg[i][j], false, false);
      }
    __syncthreads();   // buf free before it is overwritten two tiles later
  }

  // epilogue: bias + GLU, write final f32 output [B, H, L]
  const int radd = (lane < 16) ? 0 : 8;
  const int ncol = lane & 15;
#pragma unroll
  for (int i = 0; i < 2; ++i)
#pragma unroll
    for (int j = 0; j < 2; ++j) {
      const int obase = m0 + wm * 32 + i * 16;
      const int n     = n0 + wn * 32 + j * 16 + ncol;
#pragma unroll
      for (int r = 0; r < 8; ++r) {
        const int o = obase + r + radd;
        float a = ca[i][j][r] + bout[o];
        float g = cg[i][j][r] + bout[H_ + o];
        float sg = 1.f / (1.f + expf(-g));
        out[((size_t)b * H_ + o) * (size_t)L_ + n] = a * sg;
      }
    }
}

extern "C" void kernel_launch(void* const* d_in, const int* in_sizes, int n_in,
                              void* d_out, int out_size, void* d_ws, size_t ws_size,
                              hipStream_t stream) {
  const float* u    = (const float*)d_in[0];
  const float* kern = (const float*)d_in[1];
  const float* Dv   = (const float*)d_in[2];
  const float* W    = (const float*)d_in[3];
  const float* bo   = (const float*)d_in[4];
  float* out = (float*)d_out;

  __bf16* Wb = (__bf16*)d_ws;                        // 16 MiB
  __bf16* Yt = Wb + (size_t)O_ * H_;                 // 32 MiB, [B, L, H]

  wcvt_kernel<<<dim3((O_ * H_) / (256 * 4)), 256, 0, stream>>>(W, Wb);
  conv_gelu_kernel<<<dim3(L_ / LT2, H_ / HT, B_), 256, 0, stream>>>(u, kern, Dv, Yt);
  gemm_glu_kernel<<<dim3(L_ / BN, H_ / BM, B_), 256, 0, stream>>>(Wb, Yt, bo, out);
}
